// MyConditionalGlobalAttention_73907797229759
// MI455X (gfx1250) — matrix operand order repaired and checked by reading, not compiled, not run on hardware
//
#include <hip/hip_runtime.h>
#include <cstdint>

// ---- types for CDNA5 WMMA / TDM ----
typedef __attribute__((ext_vector_type(16))) __bf16 v16bf;
typedef __attribute__((ext_vector_type(8)))  __bf16 v8bf;
typedef __attribute__((ext_vector_type(8)))  float  v8f;
typedef __attribute__((ext_vector_type(4)))  unsigned int u32x4;
typedef __attribute__((ext_vector_type(4)))  int    i32x4;
typedef __attribute__((ext_vector_type(8)))  int    i32x8;

#define CDIM 128
#define TM   64          // rows per workgroup tile
#define LDA  136         // bf16 stride for A/H tile (128 + 8 pad)
#define LDW  136         // bf16 stride for W^T tile (pad baked into global image)
#define LDXC 132         // f32 stride for staged xc tile
#define WTILE_ELEMS (CDIM * LDW)            // 17408 bf16
#define WCOPY_DW    (CDIM * LDW * 2 / 4)    // 8704 dwords per weight image

// ---------------------------------------------------------------------------
// TDM: flat 1-D copy of one prepped weight image (34816 B) global -> LDS.
// D# per CDNA5 ISA: group0 = {count, lds_addr,
// global_addr, type=2}; group1 = {data_size=4B, tensor_dim0=tile_dim0=8704}.
// ---------------------------------------------------------------------------
__device__ __forceinline__ void tdm_load_weights(unsigned lds_off, const void* gsrc) {
    unsigned long long ga = (unsigned long long)(uintptr_t)gsrc;
    u32x4 g0;
    g0[0] = 1u;                                        // count=1 (valid), user mode
    g0[1] = lds_off;                                   // lds_addr (bytes)
    g0[2] = (unsigned)(ga & 0xFFFFFFFFu);              // global_addr[31:0]
    g0[3] = (unsigned)((ga >> 32) & 0x01FFFFFFu)       // global_addr[56:32]
            | (2u << 30);                              // type=2 ("image")
    i32x8 g1;
    g1[0] = (2 << 16);                                 // data_size = 4 bytes
    g1[1] = (int)((WCOPY_DW & 0xFFFF) << 16);          // tensor_dim0 lo -> bits 63:48
    g1[2] = (int)((WCOPY_DW >> 16) | (1u << 16));      // tensor_dim0 hi | tensor_dim1=1
    g1[3] = (int)(WCOPY_DW << 16);                     // tile_dim0 -> bits 127:112
    g1[4] = 0;                                         // tile_dim1=0, tile_dim2=0 (1-D)
    g1[5] = (int)WCOPY_DW;                             // tensor_dim0_stride lo
    g1[6] = 0;
    g1[7] = 0;
    i32x4 z4 = {0, 0, 0, 0};
#if defined(__clang_major__) && (__clang_major__ >= 23)
    i32x8 z8 = {0, 0, 0, 0, 0, 0, 0, 0};
    __builtin_amdgcn_tensor_load_to_lds(g0, g1, z4, z4, z8, 0);
#else
    __builtin_amdgcn_tensor_load_to_lds(g0, g1, z4, z4, 0);
#endif
}

// ---------------------------------------------------------------------------
// Pre-convert 4 weight matrices to bf16, transposed [out][in], stride LDW
// (so the global image == the LDS tile image, ready for a flat TDM copy).
// ---------------------------------------------------------------------------
__global__ void prep_weights_kernel(const float* __restrict__ W0,
                                    const float* __restrict__ W1,
                                    const float* __restrict__ W2,
                                    const float* __restrict__ W3,
                                    __bf16* __restrict__ dst) {
    int tid = blockIdx.x * blockDim.x + threadIdx.x;   // 65536 total
    int m   = tid >> 14;
    int rem = tid & 16383;
    int out = rem >> 7;
    int in  = rem & 127;
    const float* W = (m == 0) ? W0 : (m == 1) ? W1 : (m == 2) ? W2 : W3;
    dst[(size_t)m * WTILE_ELEMS + out * LDW + in] = (__bf16)W[in * CDIM + out];
}

// ---------------------------------------------------------------------------
// Segment boundaries from sorted batch: seg_start[g] = first n with batch[n]>=g
// ---------------------------------------------------------------------------
__global__ void segbounds_kernel(const int* __restrict__ batch,
                                 int* __restrict__ seg_start, int N, int G) {
    int n = blockIdx.x * blockDim.x + threadIdx.x;
    if (n > N) return;
    int bc = (n < N) ? batch[n] : G;
    int bp = (n == 0) ? -1 : batch[n - 1];
    for (int g = bp + 1; g <= bc; ++g) seg_start[g] = n;
}

// ---------------------------------------------------------------------------
// One 16-row band GEMM via v_wmma_f32_16x16x32_bf16.
// ---------------------------------------------------------------------------
__device__ __forceinline__ void gemm_band(const __bf16* __restrict__ sArow,
                                          const __bf16* __restrict__ sWp,
                                          int lane, v8f acc[8]) {
    const int l16 = lane & 15;
    const int lh  = lane >> 4;
    const __bf16* ar = sArow + (size_t)l16 * LDA;

    v16bf afr[4];
#pragma unroll
    for (int kc = 0; kc < 4; ++kc) {
        v8bf lo = *(const v8bf*)(ar + kc * 32 + lh * 8);
        v8bf hi = *(const v8bf*)(ar + kc * 32 + 16 + lh * 8);
        afr[kc] = __builtin_shufflevector(lo, hi, 0, 1, 2, 3, 4, 5, 6, 7,
                                          8, 9, 10, 11, 12, 13, 14, 15);
    }
#pragma unroll
    for (int nt = 0; nt < 8; ++nt) {
        const __bf16* wr = sWp + (size_t)(nt * 16 + l16) * LDW + lh * 8;
        v8f a = acc[nt];
#pragma unroll
        for (int kc = 0; kc < 4; ++kc) {
            v8bf lo = *(const v8bf*)(wr + kc * 32);
            v8bf hi = *(const v8bf*)(wr + kc * 32 + 16);
            v16bf bf = __builtin_shufflevector(lo, hi, 0, 1, 2, 3, 4, 5, 6, 7,
                                               8, 9, 10, 11, 12, 13, 14, 15);
            a = __builtin_amdgcn_wmma_f32_16x16x32_bf16(
                    false, afr[kc], false, bf, (short)0, a, false, false);
        }
        acc[nt] = a;
    }
}

// ---------------------------------------------------------------------------
// Fused MLP: Y = ReLU(X @ W1 + b1) @ W2 + b2 for a 64-row tile; weights come
// pre-transposed/bf16 from global and are DMA'd into LDS by the TDM.
// If gate != nullptr also writes gate[n] = dot(Y[n], q[batch[n]]).
// ---------------------------------------------------------------------------
__global__ __launch_bounds__(128) void mlp128_kernel(
    const float* __restrict__ X,
    const __bf16* __restrict__ Wt1, const float* __restrict__ b1,
    const __bf16* __restrict__ Wt2, const float* __restrict__ b2,
    float* __restrict__ Y, int M,
    const int* __restrict__ batch, const float* __restrict__ qmat,
    float* __restrict__ gate) {

    __shared__ __align__(16) __bf16 sW[WTILE_ELEMS];  // W^T tile, reused for xc stage
    __shared__ __align__(16) __bf16 sA[TM * LDA];     // X tile -> H tile

    const int t    = threadIdx.x;
    const int lane = t & 31;
    const int wv   = t >> 5;
    const int m0   = blockIdx.x * TM;
    const unsigned sW_off = (unsigned)(uintptr_t)(void*)&sW[0];

    // ---- wave 0: kick off TDM copy of W1^T into LDS (overlaps X-tile load) ----
    if (t < 32) tdm_load_weights(sW_off, Wt1);

    // ---- all threads: load X tile (f32 -> bf16) ----
#pragma unroll
    for (int it = 0; it < 16; ++it) {
        int idx = t + it * 128;            // float4 index (2048 total)
        int row = idx >> 5;                // 32 float4 per row
        int c4  = idx & 31;
        int gr  = m0 + row;
        float4 v = make_float4(0.f, 0.f, 0.f, 0.f);
        if (gr < M) v = *(const float4*)(X + (size_t)gr * CDIM + c4 * 4);
        __bf16* d = &sA[row * LDA + c4 * 4];
        d[0] = (__bf16)v.x; d[1] = (__bf16)v.y;
        d[2] = (__bf16)v.z; d[3] = (__bf16)v.w;
    }
    if (t < 32) __builtin_amdgcn_s_wait_tensorcnt(0);
    __syncthreads();

    // ---- GEMM1: H = ReLU(X @ W1 + b1) ----
    v8f acc[8] = {};
    gemm_band(&sA[wv * 16 * LDA], sW, lane, acc);

    const int l16 = lane & 15;
    const int lh  = lane >> 4;
#pragma unroll
    for (int nt = 0; nt < 8; ++nt) {
        int col  = nt * 16 + l16;
        float bb = b1[col];
#pragma unroll
        for (int r = 0; r < 8; ++r) {
            int row = wv * 16 + r + lh * 8;
            float v = acc[nt][r] + bb;
            v = v > 0.f ? v : 0.f;
            sA[row * LDA + col] = (__bf16)v;
        }
    }
    __syncthreads();                    // everyone done reading W1^T

    // ---- wave 0: TDM copy of W2^T over W1^T ----
    if (t < 32) {
        tdm_load_weights(sW_off, Wt2);
        __builtin_amdgcn_s_wait_tensorcnt(0);
    }
    __syncthreads();

    // ---- GEMM2: Y = H @ W2 + b2 ----
    v8f acc2[8] = {};
    gemm_band(&sA[wv * 16 * LDA], sW, lane, acc2);
    __syncthreads();                    // everyone done reading sW -> reuse as xc stage

    float* sXC = (float*)sW;            // 64 x LDXC floats (fits in sW)
#pragma unroll
    for (int nt = 0; nt < 8; ++nt) {
        int col  = nt * 16 + l16;
        float bb = b2[col];
#pragma unroll
        for (int r = 0; r < 8; ++r) {
            int row = wv * 16 + r + lh * 8;
            sXC[row * LDXC + col] = acc2[nt][r] + bb;
        }
    }
    __syncthreads();

    // ---- coalesced store of Y tile ----
#pragma unroll
    for (int it = 0; it < 16; ++it) {
        int idx = t + it * 128;
        int row = idx >> 5;
        int c4  = idx & 31;
        int gr  = m0 + row;
        if (gr < M) {
            const float* sp = &sXC[row * LDXC + c4 * 4];
            float4 v = make_float4(sp[0], sp[1], sp[2], sp[3]);
            *(float4*)(Y + (size_t)gr * CDIM + c4 * 4) = v;
        }
    }

    // ---- optional gate: gate[n] = dot(Y[n], q[batch[n]]) ----
    if (gate != nullptr) {
        int node = t >> 1;
        int half = t & 1;
        int gr   = m0 + node;
        float sum = 0.f;
        if (gr < M) {
            int g = batch[gr];
            const float* qr = qmat + (size_t)g * CDIM + half * 64;
            const float* xr = &sXC[node * LDXC + half * 64];
#pragma unroll 16
            for (int j = 0; j < 64; ++j) sum += xr[j] * qr[j];
        }
        sum += __shfl_xor(sum, 1, 32);
        if (half == 0 && gr < M) gate[gr] = sum;
    }
}

// ---------------------------------------------------------------------------
// Per-graph segment softmax + gated sum:  out[g] = sum_n alpha_n * xc[n]
// ---------------------------------------------------------------------------
#define MAXSEG 4096
__global__ __launch_bounds__(128) void softout_kernel(
    const float* __restrict__ xc, const float* __restrict__ gate,
    const int* __restrict__ seg_start, float* __restrict__ out) {

    __shared__ float sg[MAXSEG];
    __shared__ float red[128];

    const int g = blockIdx.x;
    const int t = threadIdx.x;
    const int s = seg_start[g];
    const int e = seg_start[g + 1];
    int len = e - s;
    if (len > MAXSEG) len = MAXSEG;     // never hit for N=1e6, G=8192
    if (len <= 0) { out[(size_t)g * CDIM + t] = 0.f; return; }

    // max
    float mx = -3.402823466e38f;
    for (int i = t; i < len; i += 128) { float v = gate[s + i]; sg[i] = v; mx = fmaxf(mx, v); }
    red[t] = mx; __syncthreads();
    for (int st = 64; st > 0; st >>= 1) {
        if (t < st) red[t] = fmaxf(red[t], red[t + st]);
        __syncthreads();
    }
    float m = red[0]; __syncthreads();

    // exp + denom
    float sum = 0.f;
    for (int i = t; i < len; i += 128) { float ev = __expf(sg[i] - m); sg[i] = ev; sum += ev; }
    red[t] = sum; __syncthreads();
    for (int st = 64; st > 0; st >>= 1) {
        if (t < st) red[t] += red[t + st];
        __syncthreads();
    }
    float scale = 1.f / (red[0] + 1e-16f);
    __syncthreads();

    // weighted sum: thread t owns column t (coalesced xc row reads)
    float acc = 0.f;
    for (int i = 0; i < len; ++i)
        acc += sg[i] * xc[(size_t)(s + i) * CDIM + t];
    out[(size_t)g * CDIM + t] = acc * scale;
}

// ---------------------------------------------------------------------------
extern "C" void kernel_launch(void* const* d_in, const int* in_sizes, int n_in,
                              void* d_out, int out_size, void* d_ws, size_t ws_size,
                              hipStream_t stream) {
    const float* x   = (const float*)d_in[0];
    const float* u   = (const float*)d_in[1];
    const float* Wn1 = (const float*)d_in[2];
    const float* bn1 = (const float*)d_in[3];
    const float* Wn2 = (const float*)d_in[4];
    const float* bn2 = (const float*)d_in[5];
    const float* Wq1 = (const float*)d_in[6];
    const float* bq1 = (const float*)d_in[7];
    const float* Wq2 = (const float*)d_in[8];
    const float* bq2 = (const float*)d_in[9];
    const int*  batch = (const int*)d_in[10];
    float* out = (float*)d_out;

    const int C = CDIM;
    const int N = in_sizes[0] / C;
    const int G = 8192;                       // reference constant ('size' lives on device)
    const int U = in_sizes[1] / (G * C);

    // workspace carve: Wt[4] | q | gate | seg_start | xc
    __bf16* wt = (__bf16*)d_ws;                              // 4 * 34816 B (256-aligned)
    float* q     = (float*)(((uintptr_t)(wt + 4 * WTILE_ELEMS) + 255u) & ~(uintptr_t)255u);
    float* gateb = q + (size_t)G * C;
    int*   seg   = (int*)(gateb + N);
    float* xc    = (float*)(((uintptr_t)(seg + (G + 1)) + 255u) & ~(uintptr_t)255u);

    prep_weights_kernel<<<256, 256, 0, stream>>>(Wn1, Wn2, Wq1, Wq2, wt);
    segbounds_kernel<<<(N + 1 + 255) / 256, 256, 0, stream>>>(batch, seg, N, G);

    const __bf16* Wn1t = wt + 0 * WTILE_ELEMS;
    const __bf16* Wn2t = wt + 1 * WTILE_ELEMS;
    const __bf16* Wq1t = wt + 2 * WTILE_ELEMS;
    const __bf16* Wq2t = wt + 3 * WTILE_ELEMS;

    for (int i = 0; i < U; ++i) {
        // q = MLP_q(u[i])           [G, C]
        mlp128_kernel<<<(G + TM - 1) / TM, 128, 0, stream>>>(
            u + (size_t)i * G * C, Wq1t, bq1, Wq2t, bq2, q, G,
            nullptr, nullptr, nullptr);
        // xc = MLP_n(xc_prev), gate = <xc, q[batch]>   [N, C], [N]
        mlp128_kernel<<<(N + TM - 1) / TM, 128, 0, stream>>>(
            (i == 0) ? x : xc, Wn1t, bn1, Wn2t, bn2, xc, N,
            batch, q, gateb);
        // out[i] = segment softmax-gated sum
        softout_kernel<<<G, 128, 0, stream>>>(xc, gateb, seg, out + (size_t)i * G * C);
    }
}
